// Decoder_RNN_Gumbel_3624952398621
// MI455X (gfx1250) — matrix operand order repaired and checked by reading, not compiled, not run on hardware
//
#include <hip/hip_runtime.h>

// ---------------------------------------------------------------------------
// Problem constants (from reference)
// ---------------------------------------------------------------------------
#define B_   64
#define L_   1024
#define KV_  256
#define H_   512
#define E_   256
#define T_   300
#define V_   33
#define MH_  512
#define H4_  2048          // 4*H
#define NEG_SLOPE_ 0.9f
#define EPS_ 1e-10f

typedef __attribute__((ext_vector_type(16))) __bf16 v16bf;
typedef __attribute__((ext_vector_type(8)))  float  v8f;

// ---------------------------------------------------------------------------
// helpers
// ---------------------------------------------------------------------------
__device__ __forceinline__ __bf16 f2bf(float f) {
  union { float f; unsigned u; } v; v.f = f;
  unsigned r = v.u + 0x7FFFu + ((v.u >> 16) & 1u);   // round-to-nearest-even
  union { unsigned short s; __bf16 b; } o;
  o.s = (unsigned short)(r >> 16);
  return o.b;
}

__device__ __forceinline__ float sigf(float x) { return 1.0f / (1.0f + __expf(-x)); }

__device__ __forceinline__ v8f wmma_bf16(v16bf a, v16bf b, v8f c) {
  return __builtin_amdgcn_wmma_f32_16x16x32_bf16(false, a, false, b, (short)0, c, false, false);
}

// A fragment: 16x32 bf16 tile of X (row-major, ld = K stride).
// Lane = row (lane&15); lanes hold K = g*8..g*8+7 and 16+g*8.., g = lane>>4.
__device__ __forceinline__ v16bf load_a_frag(const __bf16* X, int ld, int row0, int k0) {
  int lane = threadIdx.x & 31;
  int m = row0 + (lane & 15);
  int g = lane >> 4;
  const __bf16* p = X + (size_t)m * ld + k0 + g * 8;
  v16bf a;
#pragma unroll
  for (int e = 0; e < 8; ++e) a[e] = p[e];
  p += 16;
#pragma unroll
  for (int e = 0; e < 8; ++e) a[8 + e] = p[e];
  return a;
}

// B fragment: 32x16 (KxN) tile where B[k][n] = W[n*ld + k], W row-major [N,K].
// Lane = col (lane&15); lanes hold K = g*8..g*8+7 and 16+g*8.., g = lane>>4.
__device__ __forceinline__ v16bf load_b_frag(const __bf16* W, int ld, int n0, int k0) {
  int lane = threadIdx.x & 31;
  int n = n0 + (lane & 15);
  int g = lane >> 4;
  const __bf16* p = W + (size_t)n * ld + k0 + g * 8;
  v16bf b;
#pragma unroll
  for (int e = 0; e < 16; ++e) b[e] = p[e < 8 ? e : e + 8];
  return b;
}

// C/D tile store for LSTM gates (bias = bih[n] + bhh[n])
__device__ __forceinline__ void store_gates(float* __restrict__ g, int row0, int n0,
                                            v8f acc, const float* __restrict__ bih,
                                            const float* __restrict__ bhh) {
  int lane = threadIdx.x & 31;
  int n = n0 + (lane & 15);
  int mb = row0 + 8 * (lane >> 4);
  float bias = bih[n] + bhh[n];
#pragma unroll
  for (int r = 0; r < 8; ++r) g[(size_t)(mb + r) * H4_ + n] = acc[r] + bias;
}

// C/D tile store for generic GEMM (+bias, optional leaky, optional bf16 mirror ld=512)
__device__ __forceinline__ void store_out(float* __restrict__ outf, __bf16* __restrict__ outbf,
                                          int N, int row0, int n0, v8f acc,
                                          const float* __restrict__ bias, int leaky) {
  int lane = threadIdx.x & 31;
  int n = n0 + (lane & 15);
  int mb = row0 + 8 * (lane >> 4);
  float bs = bias[n];
#pragma unroll
  for (int r = 0; r < 8; ++r) {
    float v = acc[r] + bs;
    if (leaky) v = (v >= 0.0f) ? v : NEG_SLOPE_ * v;
    outf[(size_t)(mb + r) * N + n] = v;
    if (outbf) outbf[(size_t)(mb + r) * H_ + n] = f2bf(v);
  }
}

// ---------------------------------------------------------------------------
// f32 -> bf16 conversion (weights once per call, activations per producer)
// ---------------------------------------------------------------------------
__global__ void f32_to_bf16(const float* __restrict__ src, __bf16* __restrict__ dst, int n) {
  int i = blockIdx.x * blockDim.x + threadIdx.x;
  if (i < n) dst[i] = f2bf(src[i]);
}

// ---------------------------------------------------------------------------
// init: c=0, h(bf16)=0, rnn_in = [emb(SOS), zeros]  (bf16)
// ---------------------------------------------------------------------------
__global__ void init_state(float* __restrict__ c, __bf16* __restrict__ hbf,
                           __bf16* __restrict__ rnn_in_bf,
                           const float* __restrict__ W_emb, const float* __restrict__ b_emb) {
  int i = blockIdx.x * blockDim.x + threadIdx.x;
  if (i < 3 * B_ * H_) { c[i] = 0.0f; hbf[i] = f2bf(0.0f); }
  if (i < B_ * H_) {
    int col = i % H_;
    float v = 0.0f;
    if (col < E_) v = W_emb[col * V_ + 0] + b_emb[col];   // SOS one-hot through Linear
    rnn_in_bf[i] = f2bf(v);
  }
}

// labels_padded [T,B] (int) -> out [B,T] (float)
__global__ void copy_labels(const int* __restrict__ labels, float* __restrict__ out) {
  int i = blockIdx.x * blockDim.x + threadIdx.x;
  if (i < T_ * B_) {
    int t = i / B_, b = i % B_;
    out[b * T_ + t] = (float)labels[i];
  }
}

// ---------------------------------------------------------------------------
// LSTM gates GEMM: gates[64,2048] = X@Wih^T + h@Whh^T + bih + bhh
// one wave per 32x32 tile: 2x2 accumulators, each fragment reused twice
// ---------------------------------------------------------------------------
__global__ __launch_bounds__(32) void gates_gemm(
    const __bf16* __restrict__ X, const __bf16* __restrict__ Hprev,
    const __bf16* __restrict__ Wih, const __bf16* __restrict__ Whh,
    const float* __restrict__ bih, const float* __restrict__ bhh,
    float* __restrict__ gates) {
  int n0 = blockIdx.x * 32;
  int row0 = blockIdx.y * 32;
  v8f acc00 = {}, acc01 = {}, acc10 = {}, acc11 = {};
#pragma unroll
  for (int k0 = 0; k0 < H_; k0 += 32) {
    v16bf a0 = load_a_frag(X, H_, row0, k0);
    v16bf a1 = load_a_frag(X, H_, row0 + 16, k0);
    v16bf b0 = load_b_frag(Wih, H_, n0, k0);
    v16bf b1 = load_b_frag(Wih, H_, n0 + 16, k0);
    acc00 = wmma_bf16(a0, b0, acc00);
    acc01 = wmma_bf16(a0, b1, acc01);
    acc10 = wmma_bf16(a1, b0, acc10);
    acc11 = wmma_bf16(a1, b1, acc11);
  }
#pragma unroll
  for (int k0 = 0; k0 < H_; k0 += 32) {
    v16bf a0 = load_a_frag(Hprev, H_, row0, k0);
    v16bf a1 = load_a_frag(Hprev, H_, row0 + 16, k0);
    v16bf b0 = load_b_frag(Whh, H_, n0, k0);
    v16bf b1 = load_b_frag(Whh, H_, n0 + 16, k0);
    acc00 = wmma_bf16(a0, b0, acc00);
    acc01 = wmma_bf16(a0, b1, acc01);
    acc10 = wmma_bf16(a1, b0, acc10);
    acc11 = wmma_bf16(a1, b1, acc11);
  }
  store_gates(gates, row0,      n0,      acc00, bih, bhh);
  store_gates(gates, row0,      n0 + 16, acc01, bih, bhh);
  store_gates(gates, row0 + 16, n0,      acc10, bih, bhh);
  store_gates(gates, row0 + 16, n0 + 16, acc11, bih, bhh);
}

// ---------------------------------------------------------------------------
// generic GEMM: out[64,N] = A[64,K]@W[N,K]^T + bias ; optional leaky-relu;
// optional bf16 mirror into outbf (ld = 512). 32x32 tile per wave.
// ---------------------------------------------------------------------------
__global__ __launch_bounds__(32) void gemm_bias(
    const __bf16* __restrict__ A, const __bf16* __restrict__ W,
    const float* __restrict__ bias, float* __restrict__ outf,
    __bf16* __restrict__ outbf, int K, int N, int leaky) {
  int n0 = blockIdx.x * 32;
  int row0 = blockIdx.y * 32;
  v8f acc00 = {}, acc01 = {}, acc10 = {}, acc11 = {};
#pragma unroll 4
  for (int k0 = 0; k0 < K; k0 += 32) {
    v16bf a0 = load_a_frag(A, K, row0, k0);
    v16bf a1 = load_a_frag(A, K, row0 + 16, k0);
    v16bf b0 = load_b_frag(W, K, n0, k0);
    v16bf b1 = load_b_frag(W, K, n0 + 16, k0);
    acc00 = wmma_bf16(a0, b0, acc00);
    acc01 = wmma_bf16(a0, b1, acc01);
    acc10 = wmma_bf16(a1, b0, acc10);
    acc11 = wmma_bf16(a1, b1, acc11);
  }
  store_out(outf, outbf, N, row0,      n0,      acc00, bias, leaky);
  store_out(outf, outbf, N, row0,      n0 + 16, acc01, bias, leaky);
  store_out(outf, outbf, N, row0 + 16, n0,      acc10, bias, leaky);
  store_out(outf, outbf, N, row0 + 16, n0 + 16, acc11, bias, leaky);
}

// ---------------------------------------------------------------------------
// LSTM pointwise: i,f,g,o -> c,h (h kept in bf16 only; feeds next GEMMs)
// ---------------------------------------------------------------------------
__global__ void lstm_pointwise(const float* __restrict__ gates,
                               float* __restrict__ c, __bf16* __restrict__ hbf) {
  int idx = blockIdx.x * blockDim.x + threadIdx.x;
  if (idx >= B_ * H_) return;
  int b = idx / H_, j = idx % H_;
  const float* g = gates + (size_t)b * H4_;
  float ig = sigf(g[j]);
  float fg = sigf(g[H_ + j]);
  float gg = tanhf(g[2 * H_ + j]);
  float og = sigf(g[3 * H_ + j]);
  float cn = fg * c[idx] + ig * gg;
  c[idx] = cn;
  hbf[idx] = f2bf(og * tanhf(cn));
}

// ---------------------------------------------------------------------------
// attention: energy -> masked-mult softmax -> ctx ; writes attn[B,L,T] slice
// one block per batch row, 256 threads, fp32 against L2-resident enc tensors,
// with global_prefetch ahead of the streaming loops
// ---------------------------------------------------------------------------
__global__ __launch_bounds__(256) void attention_kernel(
    const float* __restrict__ enc_key, const float* __restrict__ enc_value,
    const int* __restrict__ seqlens, const float* __restrict__ query,
    float* __restrict__ ctx, __bf16* __restrict__ qc_bf,
    float* __restrict__ att_out, int t) {
  __shared__ float q[KV_];
  __shared__ float attl[L_];
  __shared__ float red[8];
  __shared__ float gval;
  int b = blockIdx.x;
  int tid = threadIdx.x;
  q[tid] = query[b * KV_ + tid];
  __syncthreads();
  int sl = seqlens[b];

  float e[4];
#pragma unroll
  for (int i = 0; i < 4; ++i) {
    int l = tid + 256 * i;
    const float* kp = enc_key + ((size_t)b * L_ + l) * KV_;
    if (i < 3)  // prefetch next row this thread will consume
      __builtin_prefetch(enc_key + ((size_t)b * L_ + l + 256) * KV_, 0, 0);
    float s = 0.0f;
    for (int k = 0; k < KV_; ++k) s += kp[k] * q[k];
    e[i] = (l < sl) ? s : 0.0f;      // multiplicative mask (faithful: energy*mask)
  }

  // block max
  float m = fmaxf(fmaxf(e[0], e[1]), fmaxf(e[2], e[3]));
#pragma unroll
  for (int off = 16; off > 0; off >>= 1) m = fmaxf(m, __shfl_xor(m, off, 32));
  if ((tid & 31) == 0) red[tid >> 5] = m;
  __syncthreads();
  if (tid == 0) {
    float v = red[0];
    for (int w = 1; w < 8; ++w) v = fmaxf(v, red[w]);
    gval = v;
  }
  __syncthreads();
  float gm = gval;
  __syncthreads();

  // exp + block sum
  float s = 0.0f;
#pragma unroll
  for (int i = 0; i < 4; ++i) { e[i] = __expf(e[i] - gm); s += e[i]; }
#pragma unroll
  for (int off = 16; off > 0; off >>= 1) s += __shfl_xor(s, off, 32);
  if ((tid & 31) == 0) red[tid >> 5] = s;
  __syncthreads();
  if (tid == 0) {
    float v = 0.0f;
    for (int w = 0; w < 8; ++w) v += red[w];
    gval = v;
  }
  __syncthreads();
  float inv = 1.0f / gval;

#pragma unroll
  for (int i = 0; i < 4; ++i) {
    int l = tid + 256 * i;
    float a = e[i] * inv;
    attl[l] = a;
    att_out[((size_t)b * L_ + l) * T_ + t] = a;   // attentions[b, l, t]
  }
  __syncthreads();

  // ctx[k] = sum_l att[l] * enc_value[b,l,k]  (k = tid), prefetch 8 rows ahead
  float cs = 0.0f;
  const float* vp = enc_value + (size_t)b * L_ * KV_ + tid;
  for (int l = 0; l < L_; ++l) {
    if (l + 8 < L_) __builtin_prefetch(vp + (size_t)(l + 8) * KV_, 0, 0);
    cs += attl[l] * vp[(size_t)l * KV_];
  }
  ctx[b * KV_ + tid] = cs;
  qc_bf[(size_t)b * H_ + KV_ + tid] = f2bf(cs);   // ctx half of [query|ctx]
}

// ---------------------------------------------------------------------------
// output head: y_t = hmid@W_mlp2^T+b ; gumbel softmax ; argmax ; emb feedback
// one block per batch row
// ---------------------------------------------------------------------------
__global__ __launch_bounds__(256) void output_kernel(
    const float* __restrict__ hmid, const float* __restrict__ W_mlp2,
    const float* __restrict__ b_mlp2, const float* __restrict__ gumbel_u,
    const float* __restrict__ W_emb, const float* __restrict__ b_emb,
    const float* __restrict__ ctx, float* __restrict__ yhat_out,
    float* __restrict__ label_out, __bf16* __restrict__ rnn_in_bf, int t) {
  __shared__ float yt[V_];
  __shared__ float ysm[V_];
  int b = blockIdx.x;
  int tid = threadIdx.x;

  if (tid < V_) {
    const float* hp = hmid + (size_t)b * MH_;
    const float* wp = W_mlp2 + (size_t)tid * MH_;
    float s = b_mlp2[tid];
    for (int k = 0; k < MH_; ++k) s += hp[k] * wp[k];
    yt[tid] = s;
    yhat_out[((size_t)b * T_ + t) * V_ + tid] = s;   // y_hat[b, t, v]
  }
  __syncthreads();

  if (tid == 0) {
    float z[V_];
    float mx = -3.4e38f;
    for (int v = 0; v < V_; ++v) {
      float u = gumbel_u[((size_t)t * B_ + b) * V_ + v];
      float g = -logf(-logf(u + EPS_) + EPS_);
      z[v] = yt[v] + g;
      mx = fmaxf(mx, z[v]);
    }
    float s = 0.0f;
    for (int v = 0; v < V_; ++v) { z[v] = __expf(z[v] - mx); s += z[v]; }
    float inv = 1.0f / s;
    int am = 0; float best = -1.0f;
    for (int v = 0; v < V_; ++v) {
      float p = z[v] * inv;
      ysm[v] = p;
      if (p > best) { best = p; am = v; }   // first-max wins, matches argmax
    }
    label_out[b * T_ + t] = (float)am;
  }
  __syncthreads();

  // emb[e] = ysm @ W_emb^T + b_emb ; feedback rnn_in = [emb | ctx]  (e = tid)
  {
    int e = tid;
    float s = b_emb[e];
    const float* we = W_emb + (size_t)e * V_;
    for (int v = 0; v < V_; ++v) s += ysm[v] * we[v];
    rnn_in_bf[(size_t)b * H_ + e] = f2bf(s);
    rnn_in_bf[(size_t)b * H_ + E_ + e] = f2bf(ctx[b * KV_ + e]);
  }
}

// ---------------------------------------------------------------------------
// launch
// ---------------------------------------------------------------------------
extern "C" void kernel_launch(void* const* d_in, const int* in_sizes, int n_in,
                              void* d_out, int out_size, void* d_ws, size_t ws_size,
                              hipStream_t stream) {
  (void)in_sizes; (void)n_in; (void)out_size; (void)ws_size;

  const float* enc_key   = (const float*)d_in[0];
  const float* enc_value = (const float*)d_in[1];
  const int*   labels    = (const int*)d_in[2];
  const int*   seqlens   = (const int*)d_in[3];
  const float* gumbel_u  = (const float*)d_in[4];
  const float* W_emb     = (const float*)d_in[5];
  const float* b_emb     = (const float*)d_in[6];
  const float* W_fc      = (const float*)d_in[7];
  const float* b_fc      = (const float*)d_in[8];
  const float* W_mlp1    = (const float*)d_in[9];
  const float* b_mlp1    = (const float*)d_in[10];
  const float* W_mlp2    = (const float*)d_in[11];
  const float* b_mlp2    = (const float*)d_in[12];
  const float *Wih[3], *Whh[3], *bih[3], *bhh[3];
  for (int l = 0; l < 3; ++l) {
    Wih[l] = (const float*)d_in[13 + 4 * l];
    Whh[l] = (const float*)d_in[14 + 4 * l];
    bih[l] = (const float*)d_in[15 + 4 * l];
    bhh[l] = (const float*)d_in[16 + 4 * l];
  }

  // outputs, concatenated flat in return order
  float* out   = (float*)d_out;
  float* yhat  = out;                                        // [B,T,V]
  float* lbl   = yhat + (size_t)B_ * T_ * V_;                // [B,T]
  float* lblT  = lbl  + (size_t)B_ * T_;                     // [B,T]
  float* attn  = lblT + (size_t)B_ * T_;                     // [B,L,T]

  // workspace carve-up
  char* p = (char*)d_ws;
  auto alloc = [&](size_t bytes) -> char* {
    char* r = p; p += (bytes + 255) & ~(size_t)255; return r;
  };
  float*  c         = (float*)alloc((size_t)3 * B_ * H_ * 4);
  float*  gates     = (float*)alloc((size_t)B_ * H4_ * 4);
  float*  query     = (float*)alloc((size_t)B_ * KV_ * 4);
  float*  ctx       = (float*)alloc((size_t)B_ * KV_ * 4);
  float*  hmid      = (float*)alloc((size_t)B_ * MH_ * 4);
  __bf16* hbf       = (__bf16*)alloc((size_t)3 * B_ * H_ * 2);
  __bf16* rnn_in_bf = (__bf16*)alloc((size_t)B_ * H_ * 2);
  __bf16* qcbf      = (__bf16*)alloc((size_t)B_ * H_ * 2);
  __bf16* Wihbf[3], *Whhbf[3];
  for (int l = 0; l < 3; ++l) {
    Wihbf[l] = (__bf16*)alloc((size_t)H4_ * H_ * 2);
    Whhbf[l] = (__bf16*)alloc((size_t)H4_ * H_ * 2);
  }
  __bf16* Wfcbf = (__bf16*)alloc((size_t)KV_ * H_ * 2);
  __bf16* Wm1bf = (__bf16*)alloc((size_t)MH_ * H_ * 2);

  // weight conversions (per call — no cross-call caching allowed)
  const int nW = H4_ * H_;
  for (int l = 0; l < 3; ++l) {
    f32_to_bf16<<<(nW + 255) / 256, 256, 0, stream>>>(Wih[l], Wihbf[l], nW);
    f32_to_bf16<<<(nW + 255) / 256, 256, 0, stream>>>(Whh[l], Whhbf[l], nW);
  }
  f32_to_bf16<<<(KV_ * H_ + 255) / 256, 256, 0, stream>>>(W_fc, Wfcbf, KV_ * H_);
  f32_to_bf16<<<(MH_ * H_ + 255) / 256, 256, 0, stream>>>(W_mlp1, Wm1bf, MH_ * H_);

  init_state<<<(3 * B_ * H_ + 255) / 256, 256, 0, stream>>>(c, hbf, rnn_in_bf, W_emb, b_emb);
  copy_labels<<<(T_ * B_ + 255) / 256, 256, 0, stream>>>(labels, lblT);

  dim3 gGate(H4_ / 32, B_ / 32);   // 64 x 2 (32x32 tiles)
  dim3 gQ(KV_ / 32, B_ / 32);      // 8  x 2
  dim3 gM1(MH_ / 32, B_ / 32);     // 16 x 2

  for (int t = 0; t < T_; ++t) {
    for (int l = 0; l < 3; ++l) {
      const __bf16* x = (l == 0) ? rnn_in_bf : (hbf + (size_t)(l - 1) * B_ * H_);
      gates_gemm<<<gGate, 32, 0, stream>>>(x, hbf + (size_t)l * B_ * H_,
                                           Wihbf[l], Whhbf[l], bih[l], bhh[l], gates);
      lstm_pointwise<<<(B_ * H_ + 255) / 256, 256, 0, stream>>>(
          gates, c + (size_t)l * B_ * H_, hbf + (size_t)l * B_ * H_);
    }
    // query = h2 @ W_fc^T + b_fc  (also mirrored bf16 into qc[:, :256])
    gemm_bias<<<gQ, 32, 0, stream>>>(hbf + (size_t)2 * B_ * H_, Wfcbf, b_fc,
                                     query, qcbf, H_, KV_, 0);
    attention_kernel<<<B_, 256, 0, stream>>>(enc_key, enc_value, seqlens, query,
                                             ctx, qcbf, attn, t);
    // hmid = leaky([query|ctx] @ W_mlp1^T + b_mlp1)
    gemm_bias<<<gM1, 32, 0, stream>>>(qcbf, Wm1bf, b_mlp1, hmid, nullptr, H_, MH_, 1);
    output_kernel<<<B_, 256, 0, stream>>>(hmid, W_mlp2, b_mlp2, gumbel_u, W_emb,
                                          b_emb, ctx, yhat, lbl, rnn_in_bf, t);
  }
}
